// WordDecoderARNN_59871844106643
// MI455X (gfx1250) — matrix operand order repaired and checked by reading, not compiled, not run on hardware
//
#include <hip/hip_runtime.h>

typedef __attribute__((ext_vector_type(16))) _Float16     v16h;
typedef __attribute__((ext_vector_type(8)))  _Float16     v8h;
typedef __attribute__((ext_vector_type(8)))  float        v8f;
typedef __attribute__((ext_vector_type(4)))  unsigned int v4u;
typedef __attribute__((ext_vector_type(8)))  int          v8i;
typedef __attribute__((ext_vector_type(4)))  int          v4i;

#define B_     64
#define TENC   512
#define ENC_   512
#define H_     512
#define ATTN_  256
#define VOCAB_ 8000
#define MID_   512

// ---------------------------------------------------------------------------
// WMMA fragment helpers (CDNA5 v_wmma_f32_16x16x32_f16, wave32)
// A 16x32 f16: lane L row (L&15); half j -> K = kb + j,
//              kb = k0 + (lane>=16 ? 16 : 0).
// B 32x16 f16 (KxN): lane L col (L&15); half j -> K = k0 + (lane>=16?16:0)+j
//              (16 contiguous halves = one 32B load).
// C/D v8f: element (m0 + r + (lane>=16?8:0), n0 + (lane&15)) = c[r].
// ---------------------------------------------------------------------------
__device__ __forceinline__ v16h load_a_f32(const float* __restrict__ A, int lda,
                                           int m0, int k0, int lane) {
  const int row = m0 + (lane & 15);
  const int kb  = k0 + ((lane & 16) ? 16 : 0);
  const float* p = A + (size_t)row * lda + kb;
  v16h a;
#pragma unroll
  for (int j = 0; j < 8; ++j) a[j] = (_Float16)p[j];
#pragma unroll
  for (int j = 0; j < 8; ++j) a[8 + j] = (_Float16)p[j + 8];
  return a;
}

__device__ __forceinline__ v16h load_a_f16(const _Float16* __restrict__ A,
                                           int lda, int m0, int k0, int lane) {
  const _Float16* p =
      A + (size_t)(m0 + (lane & 15)) * lda + k0 + ((lane & 16) ? 16 : 0);
  v8h lo = *(const v8h*)p;
  v8h hi = *(const v8h*)(p + 8);
  v16h a;
#pragma unroll
  for (int j = 0; j < 8; ++j) {
    a[j] = lo[j];
    a[8 + j] = hi[j];
  }
  return a;
}

__device__ __forceinline__ v16h load_b_f16(const _Float16* __restrict__ W,
                                           int ldw, int n0, int k0, int lane) {
  const _Float16* p =
      W + (size_t)(n0 + (lane & 15)) * ldw + k0 + ((lane & 16) ? 16 : 0);
  return *(const v16h*)p;  // 32B aligned: ldw mult of 32, k0 mult of 32
}

__device__ __forceinline__ v8f wmma16(v16h a, v16h b, v8f c) {
  return __builtin_amdgcn_wmma_f32_16x16x32_f16(false, a, false, b, (short)0, c,
                                                false, false);
}

// ---------------------------------------------------------------------------
// Tensor Data Mover: 2D tile (tile_d1 rows x tile_d0 elems of 2B) -> LDS.
// D# per ISA 8.3/8.4: group0 {count, lds_addr, global_addr, type=2},
// group1 {data_size=1(2B), tensor dims, tile dims, dim0 stride}. Groups 2/3
// zero (<=2D tensor). Tracked with TENSORcnt.
// ---------------------------------------------------------------------------
__device__ __forceinline__ void tdm_load_2d_f16(unsigned lds_addr,
                                                unsigned long long gaddr,
                                                unsigned tile_d0,
                                                unsigned tile_d1,
                                                unsigned tensor_d0,
                                                unsigned tensor_d1,
                                                unsigned stride0) {
  v4u g0;
  g0[0] = 1u;  // count=1, user descriptor, no gather
  g0[1] = lds_addr;
  g0[2] = (unsigned)(gaddr & 0xFFFFFFFFu);
  g0[3] = (unsigned)((gaddr >> 32) & 0x1FFFFFFu) | (2u << 30);  // type=2
  v8i g1;
  g1[0] = 0x00010000;  // wg_mask=0, data_size=1 (2 bytes), no flags
  g1[1] = (int)(tensor_d0 << 16);                         // tensor_dim0[15:0]
  g1[2] = (int)((tensor_d0 >> 16) | (tensor_d1 << 16));   // d0 hi | d1 lo
  g1[3] = (int)((tensor_d1 >> 16) | (tile_d0 << 16));     // d1 hi | tile0
  g1[4] = (int)(tile_d1 & 0xFFFFu);                       // tile1 (tile2=0)
  g1[5] = (int)stride0;                                   // dim0 stride lo
  g1[6] = 0;
  g1[7] = 0;
  v4i z4 = {0, 0, 0, 0};
#if __clang_major__ >= 23
  v8i z8 = {0, 0, 0, 0, 0, 0, 0, 0};
  __builtin_amdgcn_tensor_load_to_lds(g0, g1, z4, z4, z8, 0);
#else
  __builtin_amdgcn_tensor_load_to_lds(g0, g1, z4, z4, 0);
#endif
}

// ---------------------------------------------------------------------------
// K0a: zero init.  K0b: fp32 -> f16 weight conversion (run once per call).
// ---------------------------------------------------------------------------
__global__ void zero_kernel(float* p, int n) {
  int i = blockIdx.x * 256 + threadIdx.x;
  if (i < n) p[i] = 0.f;
}

__global__ void cvt_kernel(const float* __restrict__ src,
                           _Float16* __restrict__ dst, int n) {
  int i = blockIdx.x * 256 + threadIdx.x;
  if (i < n) dst[i] = (_Float16)src[i];
}

// ---------------------------------------------------------------------------
// K1: enc_proj[b,t,a] = sum_k enc[b,t,k] * Wv[a,k]   (32768 x 256, K=512)
// grid (256,4), block 256. WG tile 128x64; wave tile 32x32 (2x2 wmma frags).
// ---------------------------------------------------------------------------
__global__ void enc_proj_kernel(const float* __restrict__ enc,
                                const _Float16* __restrict__ wv16,
                                float* __restrict__ encp) {
  const int lane = threadIdx.x & 31;
  const int wave = threadIdx.x >> 5;
  const int m_base = blockIdx.x * 128 + (wave & 3) * 32;
  const int n_base = blockIdx.y * 64 + (wave >> 2) * 32;
  v8f c[2][2] = {};
  for (int k0 = 0; k0 < ENC_; k0 += 32) {
    v16h a0 = load_a_f32(enc, ENC_, m_base, k0, lane);
    v16h a1 = load_a_f32(enc, ENC_, m_base + 16, k0, lane);
    v16h b0 = load_b_f16(wv16, ENC_, n_base, k0, lane);
    v16h b1 = load_b_f16(wv16, ENC_, n_base + 16, k0, lane);
    c[0][0] = wmma16(a0, b0, c[0][0]);
    c[0][1] = wmma16(a0, b1, c[0][1]);
    c[1][0] = wmma16(a1, b0, c[1][0]);
    c[1][1] = wmma16(a1, b1, c[1][1]);
  }
  const int rr = (lane & 16) ? 8 : 0, nn = lane & 15;
#pragma unroll
  for (int mt = 0; mt < 2; ++mt)
#pragma unroll
    for (int nt = 0; nt < 2; ++nt)
#pragma unroll
      for (int r = 0; r < 8; ++r) {
        int row = m_base + mt * 16 + r + rr;
        int col = n_base + nt * 16 + nn;
        encp[(size_t)row * ATTN_ + col] = c[mt][nt][r];
      }
}

// ---------------------------------------------------------------------------
// K2: fused GRU cell. grid 16 (32 cols of H each), block 256.
// waves 0-3: gi = ctx16 @ W_ih^T (3 gates), waves 4-7: gh = h16 @ W_hh^T.
// All GEMM operands f16 (no converts in the hot loop); fp32 h kept for the
// elementwise recurrence. Stage via LDS, then gates -> h_out (+f16 copy).
// ---------------------------------------------------------------------------
__global__ void gru_kernel(const _Float16* __restrict__ ctx16,
                           const float* __restrict__ h_in,
                           const _Float16* __restrict__ h16_in,
                           const _Float16* __restrict__ wih16,
                           const _Float16* __restrict__ whh16,
                           const float* __restrict__ b_ih,
                           const float* __restrict__ b_hh,
                           float* __restrict__ h_out,
                           _Float16* __restrict__ h16_out) {
  __shared__ float g[2][3][64][32];  // 48 KB
  const int lane = threadIdx.x & 31;
  const int wave = threadIdx.x >> 5;
  const int j0 = blockIdx.x * 32;
  const int sel = wave >> 2;  // 0 = input path, 1 = hidden path
  const int m0 = (wave & 3) * 16;
  const _Float16* A = sel ? h16_in : ctx16;
  const _Float16* W = sel ? whh16 : wih16;
  v8f c[3][2] = {};
  for (int k0 = 0; k0 < H_; k0 += 32) {
    v16h a = load_a_f16(A, H_, m0, k0, lane);
#pragma unroll
    for (int gt = 0; gt < 3; ++gt)
#pragma unroll
      for (int s = 0; s < 2; ++s) {
        v16h b = load_b_f16(W, H_, gt * H_ + j0 + s * 16, k0, lane);
        c[gt][s] = wmma16(a, b, c[gt][s]);
      }
  }
  const int rr = (lane & 16) ? 8 : 0, nn = lane & 15;
#pragma unroll
  for (int gt = 0; gt < 3; ++gt)
#pragma unroll
    for (int s = 0; s < 2; ++s)
#pragma unroll
      for (int r = 0; r < 8; ++r)
        g[sel][gt][m0 + r + rr][s * 16 + nn] = c[gt][s][r];
  __syncthreads();
  for (int i = threadIdx.x; i < 64 * 32; i += 256) {
    int m = i >> 5, nl = i & 31, col = j0 + nl;
    float ir = g[0][0][m][nl] + b_ih[col];
    float iz = g[0][1][m][nl] + b_ih[H_ + col];
    float in_ = g[0][2][m][nl] + b_ih[2 * H_ + col];
    float hr = g[1][0][m][nl] + b_hh[col];
    float hz = g[1][1][m][nl] + b_hh[H_ + col];
    float hn = g[1][2][m][nl] + b_hh[2 * H_ + col];
    float r = 1.f / (1.f + __expf(-(ir + hr)));
    float z = 1.f / (1.f + __expf(-(iz + hz)));
    float n = tanhf(in_ + r * hn);
    float hp = h_in[m * H_ + col];
    float hv = (1.f - z) * n + z * hp;
    h_out[m * H_ + col] = hv;
    h16_out[m * H_ + col] = (_Float16)hv;
  }
}

// ---------------------------------------------------------------------------
// K3: location-aware attention, one WG per batch row (fp32 VALU; M=1 matvecs).
// Context emitted f16-only: it is consumed solely as WMMA A-fragments.
// ---------------------------------------------------------------------------
__global__ void attn_kernel(const float* __restrict__ h_new,
                            const float* __restrict__ encp,
                            const float* __restrict__ enc,
                            const float* __restrict__ Wq,
                            const float* __restrict__ conv_w,
                            const float* __restrict__ conv_b,
                            const float* __restrict__ attn_b,
                            const float* __restrict__ w_score,
                            float* __restrict__ attn_out,
                            _Float16* __restrict__ ctx16_out, int t, int L) {
  const int b = blockIdx.x;
  const int tid = threadIdx.x;
  __shared__ float hrow[H_];
  __shared__ float qp[ATTN_];
  __shared__ float sc[TENC];
  __shared__ float cw0[ATTN_], cw1[ATTN_], cw2[ATTN_], cb[ATTN_], wsc[ATTN_];
  __shared__ float red[256];
  for (int i = tid; i < H_; i += 256) hrow[i] = h_new[b * H_ + i];
  cw0[tid] = conv_w[tid * 3 + 0];
  cw1[tid] = conv_w[tid * 3 + 1];
  cw2[tid] = conv_w[tid * 3 + 2];
  cb[tid] = conv_b[tid] + attn_b[tid];
  wsc[tid] = w_score[tid];
  __syncthreads();
  {  // q_proj: tid == attention feature index (ATTN == blockDim)
    float acc = 0.f;
    const float* w = Wq + (size_t)tid * H_;
    for (int k = 0; k < H_; ++k) acc += hrow[k] * w[k];
    qp[tid] = acc;
  }
  __syncthreads();
  const float* aprev = attn_out + ((size_t)b * L + (t - 1)) * TENC;  // t>0 only
  for (int tt = tid; tt < TENC; tt += 256) {
    float am1 = (t > 0 && tt > 0) ? aprev[tt - 1] : 0.f;
    float ac = (t > 0) ? aprev[tt] : 0.f;
    float ap1 = (t > 0 && tt < TENC - 1) ? aprev[tt + 1] : 0.f;
    const float* ep = encp + ((size_t)b * TENC + tt) * ATTN_;
    float s = 0.f;
    for (int a = 0; a < ATTN_; ++a) {
      float e =
          qp[a] + ep[a] + cw0[a] * am1 + cw1[a] * ac + cw2[a] * ap1 + cb[a];
      s += wsc[a] * tanhf(e);
    }
    sc[tt] = s;
  }
  __syncthreads();
  red[tid] = fmaxf(sc[tid], sc[tid + 256]);
  __syncthreads();
  for (int st = 128; st > 0; st >>= 1) {
    if (tid < st) red[tid] = fmaxf(red[tid], red[tid + st]);
    __syncthreads();
  }
  float mx = red[0];
  __syncthreads();
  float e0 = __expf(sc[tid] - mx), e1 = __expf(sc[tid + 256] - mx);
  red[tid] = e0 + e1;
  __syncthreads();
  for (int st = 128; st > 0; st >>= 1) {
    if (tid < st) red[tid] += red[tid + st];
    __syncthreads();
  }
  float inv = 1.f / red[0];
  __syncthreads();
  float a0v = e0 * inv, a1v = e1 * inv;
  sc[tid] = a0v;
  sc[tid + 256] = a1v;
  float* aout = attn_out + ((size_t)b * L + t) * TENC;
  aout[tid] = a0v;
  aout[tid + 256] = a1v;
  __syncthreads();
  float acc0 = 0.f, acc1 = 0.f;
  for (int tt = 0; tt < TENC; ++tt) {
    float a = sc[tt];
    const float* er = enc + ((size_t)b * TENC + tt) * ENC_;
    acc0 += a * er[tid];
    acc1 += a * er[tid + 256];
  }
  ctx16_out[b * ENC_ + tid] = (_Float16)acc0;
  ctx16_out[b * ENC_ + tid + 256] = (_Float16)acc1;
}

// ---------------------------------------------------------------------------
// K4: mid16 = f16(tanh(concat(h16, ctx16) @ fc1_w^T + b)). 64x512, K=1024.
// ---------------------------------------------------------------------------
__global__ void fc1_kernel(const _Float16* __restrict__ h16,
                           const _Float16* __restrict__ ctx16,
                           const _Float16* __restrict__ fc1w16,
                           const float* __restrict__ fc1_b,
                           _Float16* __restrict__ mid16) {
  const int lane = threadIdx.x & 31, wave = threadIdx.x >> 5;
  const int m0 = (wave & 3) * 16;
  const int nb = blockIdx.x * 64 + (wave >> 2) * 32;
  v8f c[2] = {};
  for (int k0 = 0; k0 < H_ + ENC_; k0 += 32) {
    const _Float16* A = (k0 < H_) ? h16 : ctx16;
    int ka = (k0 < H_) ? k0 : (k0 - H_);
    v16h a = load_a_f16(A, H_, m0, ka, lane);
#pragma unroll
    for (int s = 0; s < 2; ++s) {
      v16h bfr = load_b_f16(fc1w16, H_ + ENC_, nb + s * 16, k0, lane);
      c[s] = wmma16(a, bfr, c[s]);
    }
  }
  const int rr = (lane & 16) ? 8 : 0, nn = lane & 15;
#pragma unroll
  for (int s = 0; s < 2; ++s)
#pragma unroll
    for (int r = 0; r < 8; ++r) {
      int row = m0 + r + rr, col = nb + s * 16 + nn;
      mid16[row * MID_ + col] = (_Float16)tanhf(c[s][r] + fc1_b[col]);
    }
}

// ---------------------------------------------------------------------------
// K5: logits = mid @ fc2_w^T + b. 64x8000, K=512. grid 50 (160 cols/WG).
// TDM double-buffers the shared 160x32-half B tile into LDS (wave 0 issues,
// waits TENSORcnt, barrier); all 8 waves consume B-frags from LDS — removes
// the 4x duplicate B traffic and overlaps DMA with WMMA.
// ---------------------------------------------------------------------------
__global__ void fc2_kernel(const _Float16* __restrict__ mid16,
                           const _Float16* __restrict__ fc2w16,
                           const float* __restrict__ fc2_b,
                           float* __restrict__ pred, int t, int L) {
  __shared__ _Float16 stage[2][160 * 32];  // 20 KB double buffer
  const int lane = threadIdx.x & 31, wave = threadIdx.x >> 5;
  const int m0 = (wave & 3) * 16;
  const int nhalf = wave >> 2;  // 0 or 1
  const int colbase = blockIdx.x * 160;
  const unsigned lds0 = (unsigned)(uintptr_t)&stage[0][0];
  const unsigned long long gbase =
      (unsigned long long)(uintptr_t)fc2w16 + (size_t)colbase * MID_ * 2;

  v8f c[5] = {};
  if (wave == 0) {
    tdm_load_2d_f16(lds0, gbase, 32, 160, MID_, VOCAB_, MID_);
    __builtin_amdgcn_s_wait_tensorcnt(0);
  }
  __syncthreads();
  for (int kt = 0; kt < MID_ / 32; ++kt) {
    if (wave == 0 && kt + 1 < MID_ / 32)
      tdm_load_2d_f16(lds0 + (unsigned)(((kt + 1) & 1) * 160 * 32 * 2),
                      gbase + (unsigned long long)(kt + 1) * 32 * 2, 32, 160,
                      MID_, VOCAB_, MID_);
    const int cur = kt & 1;
    v16h a = load_a_f16(mid16, MID_, m0, kt * 32, lane);
    const int kb = (lane & 16) ? 16 : 0;
#pragma unroll
    for (int i = 0; i < 5; ++i) {
      const int nloc = ((nhalf + 2 * i) * 16 + (lane & 15)) * 32 + kb;
      v16h bfr;
#pragma unroll
      for (int j = 0; j < 16; ++j) bfr[j] = stage[cur][nloc + j];
      c[i] = wmma16(a, bfr, c[i]);
    }
    if (wave == 0 && kt + 1 < MID_ / 32) __builtin_amdgcn_s_wait_tensorcnt(0);
    __syncthreads();
  }
  const int rr = (lane & 16) ? 8 : 0, nn = lane & 15;
#pragma unroll
  for (int i = 0; i < 5; ++i)
#pragma unroll
    for (int r = 0; r < 8; ++r) {
      int brow = m0 + r + rr;
      int col = colbase + (nhalf + 2 * i) * 16 + nn;
      pred[((size_t)brow * L + t) * VOCAB_ + col] = c[i][r] + fc2_b[col];
    }
}

// ---------------------------------------------------------------------------
// K6: in-place log_softmax over VOCAB per row. grid 64.
// ---------------------------------------------------------------------------
__global__ void lsm_kernel(float* __restrict__ pred, int t, int L) {
  const int b = blockIdx.x, tid = threadIdx.x;
  float* row = pred + ((size_t)b * L + t) * VOCAB_;
  __shared__ float red[256];
  float mx = -3.4e38f;
  for (int i = tid; i < VOCAB_; i += 256) mx = fmaxf(mx, row[i]);
  red[tid] = mx;
  __syncthreads();
  for (int st = 128; st > 0; st >>= 1) {
    if (tid < st) red[tid] = fmaxf(red[tid], red[tid + st]);
    __syncthreads();
  }
  mx = red[0];
  __syncthreads();
  float s = 0.f;
  for (int i = tid; i < VOCAB_; i += 256) s += __expf(row[i] - mx);
  red[tid] = s;
  __syncthreads();
  for (int st = 128; st > 0; st >>= 1) {
    if (tid < st) red[tid] += red[tid + st];
    __syncthreads();
  }
  float lse = mx + __logf(red[0]);
  __syncthreads();
  for (int i = tid; i < VOCAB_; i += 256) row[i] = row[i] - lse;
}

// ---------------------------------------------------------------------------
extern "C" void kernel_launch(void* const* d_in, const int* in_sizes, int n_in,
                              void* d_out, int out_size, void* d_ws,
                              size_t ws_size, hipStream_t stream) {
  (void)in_sizes; (void)n_in; (void)ws_size;
  const float* enc     = (const float*)d_in[0];
  const float* W_ih    = (const float*)d_in[1];
  const float* W_hh    = (const float*)d_in[2];
  const float* b_ih    = (const float*)d_in[3];
  const float* b_hh    = (const float*)d_in[4];
  const float* Wq      = (const float*)d_in[5];
  const float* Wv      = (const float*)d_in[6];
  const float* conv_w  = (const float*)d_in[7];
  const float* conv_b  = (const float*)d_in[8];
  const float* attn_b  = (const float*)d_in[9];
  const float* w_score = (const float*)d_in[10];
  const float* fc1_w   = (const float*)d_in[11];
  const float* fc1_b   = (const float*)d_in[12];
  const float* fc2_w   = (const float*)d_in[13];
  const float* fc2_b   = (const float*)d_in[14];
  // max_length is a device scalar; derive L from out_size (graph-capture safe)
  const int L = out_size / (B_ * (VOCAB_ + TENC));

  char* ws = (char*)d_ws;
  float*     encp   = (float*)ws;                  // 33,554,432 B
  float*     h0     = (float*)(ws + 33554432);     //    131,072 B \  zeroed
  _Float16*  h16_0  = (_Float16*)(ws + 33685504);  //     65,536 B  | together
  _Float16*  ctx16  = (_Float16*)(ws + 33751040);  //     65,536 B /
  float*     h1     = (float*)(ws + 33816576);     //    131,072 B
  _Float16*  h16_1  = (_Float16*)(ws + 33947648);  //     65,536 B
  _Float16*  mid16  = (_Float16*)(ws + 34013184);  //     65,536 B
  _Float16*  wv16   = (_Float16*)(ws + 34078720);  //    262,144 B
  _Float16*  wih16  = (_Float16*)(ws + 34340864);  //  1,572,864 B
  _Float16*  whh16  = (_Float16*)(ws + 35913728);  //  1,572,864 B
  _Float16*  fc1w16 = (_Float16*)(ws + 37486592);  //  1,048,576 B
  _Float16*  fc2w16 = (_Float16*)(ws + 38535168);  //  8,192,000 B

  float* pred = (float*)d_out;
  float* attn_out = pred + (size_t)B_ * L * VOCAB_;

  // zero h0 + h16_0 + ctx16 (contiguous 262,144 B = 65,536 floats)
  zero_kernel<<<256, 256, 0, stream>>>(h0, 65536);
  cvt_kernel<<<(ATTN_ * ENC_ + 255) / 256, 256, 0, stream>>>(Wv, wv16,
                                                             ATTN_ * ENC_);
  cvt_kernel<<<(3 * H_ * ENC_ + 255) / 256, 256, 0, stream>>>(W_ih, wih16,
                                                              3 * H_ * ENC_);
  cvt_kernel<<<(3 * H_ * H_ + 255) / 256, 256, 0, stream>>>(W_hh, whh16,
                                                            3 * H_ * H_);
  cvt_kernel<<<(MID_ * (H_ + ENC_) + 255) / 256, 256, 0, stream>>>(
      fc1_w, fc1w16, MID_ * (H_ + ENC_));
  cvt_kernel<<<(VOCAB_ * MID_ + 255) / 256, 256, 0, stream>>>(fc2_w, fc2w16,
                                                              VOCAB_ * MID_);
  enc_proj_kernel<<<dim3(256, 4), 256, 0, stream>>>(enc, wv16, encp);
  for (int t = 0; t < L; ++t) {
    float*    h_in    = (t & 1) ? h1 : h0;
    float*    h_out   = (t & 1) ? h0 : h1;
    _Float16* h16_in  = (t & 1) ? h16_1 : h16_0;
    _Float16* h16_out = (t & 1) ? h16_0 : h16_1;
    gru_kernel<<<16, 256, 0, stream>>>(ctx16, h_in, h16_in, wih16, whh16, b_ih,
                                       b_hh, h_out, h16_out);
    attn_kernel<<<64, 256, 0, stream>>>(h_out, encp, enc, Wq, conv_w, conv_b,
                                        attn_b, w_score, attn_out, ctx16, t, L);
    fc1_kernel<<<8, 256, 0, stream>>>(h16_out, ctx16, fc1w16, fc1_b, mid16);
    fc2_kernel<<<50, 256, 0, stream>>>(mid16, fc2w16, fc2_b, pred, t, L);
    lsm_kernel<<<64, 256, 0, stream>>>(pred, t, L);
  }
}
